// AttnBlock_4269197492248
// MI455X (gfx1250) — matrix operand order repaired and checked by reading, not compiled, not run on hardware
//
#include <hip/hip_runtime.h>
#include <hip/hip_bf16.h>

// MI455X / gfx1250 attention block.
// f16 WMMA (v_wmma_f32_16x16x32_f16) for all 6 GEMMs, f32 for stats/softmax.
// TDM (tensor_load_to_lds + s_wait_tensorcnt) stages softmax rows into LDS.

typedef __attribute__((ext_vector_type(16))) _Float16 v16h;
typedef __attribute__((ext_vector_type(8)))  _Float16 v8h;
typedef __attribute__((ext_vector_type(8)))  float    v8f;
typedef __attribute__((ext_vector_type(4)))  unsigned int v4u;
typedef __attribute__((ext_vector_type(8)))  unsigned int v8u;

#define B_  4
#define C_  512
#define HW_ 4096
#define G_  32

// ---------------------------------------------------------------------------
// GroupNorm: one workgroup per (batch, group). 16 channels x 4096 pixels.
// ---------------------------------------------------------------------------
__global__ __launch_bounds__(1024) void groupnorm_kernel(
    const float* __restrict__ x, const float* __restrict__ gamma,
    const float* __restrict__ beta, _Float16* __restrict__ hn)
{
  const int b   = blockIdx.x / G_;
  const int g   = blockIdx.x % G_;
  const int tid = threadIdx.x;
  const size_t base = ((size_t)b * C_ + (size_t)g * 16) * HW_;
  const float* xp = x + base;

  float s = 0.f, ss = 0.f;
  for (int i = tid; i < 16 * HW_; i += 1024) {
    float v = xp[i];
    s += v; ss += v * v;
  }
  __shared__ float rs[1024], rq[1024];
  rs[tid] = s; rq[tid] = ss;
  __syncthreads();
  for (int off = 512; off > 0; off >>= 1) {
    if (tid < off) { rs[tid] += rs[tid + off]; rq[tid] += rq[tid + off]; }
    __syncthreads();
  }
  const float mean = rs[0] * (1.0f / 65536.0f);
  const float var  = rq[0] * (1.0f / 65536.0f) - mean * mean;
  const float inv  = rsqrtf(var + 1e-6f);

  _Float16* hp = hn + base;
  for (int i = tid; i < 16 * HW_; i += 1024) {
    const int c = g * 16 + (i >> 12);
    const float v = (xp[i] - mean) * inv * gamma[c] + beta[c];
    hp[i] = (_Float16)v;
  }
}

// ---------------------------------------------------------------------------
// Row softmax over scores [4096 x 4096], in place (f16 storage, f32 math).
// The 8 KB row is DMA'd into LDS by the Tensor Data Mover (one wave issues;
// TDM ignores EXEC), drained with s_wait_tensorcnt before the barrier.
// ---------------------------------------------------------------------------
__global__ __launch_bounds__(256) void softmax_kernel(_Float16* __restrict__ s)
{
  const int tid = threadIdx.x;
  _Float16* row = s + (size_t)blockIdx.x * HW_;

  __shared__ __align__(16) _Float16 hbuf[HW_];  // 8 KB
  __shared__ float ebuf[HW_];                   // 16 KB
  __shared__ float red[256];

  if (tid == 0) {
    const unsigned long long ga = (unsigned long long)(uintptr_t)row;
    const unsigned int lds = (unsigned int)(uintptr_t)hbuf;  // low 32 = LDS offset
    // D# Group 0: count=1 | lds_addr | global_addr[56:0] | type=2
    v4u g0;
    g0.x = 1u;
    g0.y = lds;
    g0.z = (unsigned int)ga;
    g0.w = (unsigned int)((ga >> 32) & 0x01FFFFFFu) | (2u << 30);
    // D# Group 1: wg_mask=0, data_size=1(2B); tensor_dim0=4096 @bits79:48;
    // tensor_dim1=1 @111:80; tile_dim0=4096 @127:112; tile_dim1=1 @143:128;
    // tensor_dim0_stride=4096 @207:160.
    v8u g1;
    g1[0] = 0x00010000u;
    g1[1] = ((unsigned int)HW_) << 16;
    g1[2] = 0x00010000u;
    g1[3] = ((unsigned int)HW_) << 16;
    g1[4] = 0x00000001u;
    g1[5] = (unsigned int)HW_;
    g1[6] = 0u;
    g1[7] = 0u;
    asm volatile("tensor_load_to_lds %0, %1" ::"s"(g0), "s"(g1) : "memory");
    __builtin_amdgcn_s_wait_tensorcnt(0);
  }
  __syncthreads();

  float mx = -3.0e38f;
  for (int i = tid; i < HW_; i += 256) mx = fmaxf(mx, (float)hbuf[i]);
  red[tid] = mx; __syncthreads();
  for (int off = 128; off > 0; off >>= 1) {
    if (tid < off) red[tid] = fmaxf(red[tid], red[tid + off]);
    __syncthreads();
  }
  mx = red[0]; __syncthreads();

  float sum = 0.f;
  for (int i = tid; i < HW_; i += 256) {
    float e = __expf((float)hbuf[i] - mx);
    ebuf[i] = e; sum += e;
  }
  red[tid] = sum; __syncthreads();
  for (int off = 128; off > 0; off >>= 1) {
    if (tid < off) red[tid] += red[tid + off];
    __syncthreads();
  }
  const float rinv = 1.0f / red[0];
  for (int i = tid; i < HW_; i += 256) row[i] = (_Float16)(ebuf[i] * rinv);
}

// ---------------------------------------------------------------------------
// WMMA GEMM: D[M x Ncols] = A[M x K] * B[K x Ncols] (+ epilogue per mode).
//  MODE_PROJ  : A = f32 weights (direct), B = f16 (direct), out f16, (d+bias)*scale
//  MODE_SCORES: A = f16 transposed-on-load (q^T), B = f16 (direct), out f16
//  MODE_PV    : A = f16 (direct, V), B = f16 transposed-on-load (P^T), out f16
//  MODE_OUT   : A = f32 weights, B = f16, out f32 = d + bias + residual
// Workgroup tile 64x128, K-step 32 (== one v_wmma_f32_16x16x32_f16).
// 8 waves (wave32) in 2x4 grid; each wave: 2x2 tiles of 16x16 -> 4 x v8f acc.
// ---------------------------------------------------------------------------
enum { MODE_PROJ = 0, MODE_SCORES = 1, MODE_PV = 2, MODE_OUT = 3 };

#define BLK_M 64
#define BLK_N 128
#define BLK_K 32
#define ASTR  40    // padded row (halves): 80 B, 16-B aligned rows
#define BSTR  136   // padded row (halves): 272 B, 16-B aligned rows

template <int MODE>
__global__ __launch_bounds__(256) void wmma_gemm_kernel(
    const float* __restrict__ Af, const _Float16* __restrict__ Ah,
    const _Float16* __restrict__ Bh, const float* __restrict__ bias,
    const float* __restrict__ resid, _Float16* __restrict__ outh,
    float* __restrict__ outf, float scale, int M, int Ncols, int K,
    int srcStride)
{
  const int tid  = threadIdx.x;
  const int m0   = blockIdx.y * BLK_M;
  const int n0   = blockIdx.x * BLK_N;

  __shared__ __align__(16) _Float16 As[BLK_M * ASTR];
  __shared__ __align__(16) _Float16 Bs[BLK_K * BSTR];

  const int lane = tid & 31;
  const int wave = tid >> 5;
  const int wm   = wave & 1;   // 0..1 -> 32-row strip
  const int wn   = wave >> 1;  // 0..3 -> 32-col strip
  const int half = lane >> 4;  // 0..1
  const int l16  = lane & 15;

  v8f acc[2][2] = {};

  for (int kb = 0; kb < K; kb += BLK_K) {
    __syncthreads();

    // ---- stage A tile (64 x 32) into LDS ----
    if constexpr (MODE == MODE_PROJ || MODE == MODE_OUT) {
      const int i = tid * 8, m = i >> 5, k = i & 31;
      const float* src = Af + (size_t)(m0 + m) * K + kb + k;
      _Float16* dst = &As[m * ASTR + k];
#pragma unroll
      for (int u = 0; u < 8; ++u) dst[u] = (_Float16)src[u];
      if (kb + BLK_K < K) __builtin_prefetch(src + BLK_K, 0, 1);
    } else if constexpr (MODE == MODE_PV) {
      const int i = tid * 8, m = i >> 5, k = i & 31;
      const _Float16* src = Ah + (size_t)(m0 + m) * K + kb + k;
      _Float16* dst = &As[m * ASTR + k];
#pragma unroll
      for (int u = 0; u < 8; ++u) dst[u] = src[u];
      if (kb + BLK_K < K) __builtin_prefetch(src + BLK_K, 0, 1);
    } else {  // MODE_SCORES: A = q^T; q stored [K][M'] with row stride srcStride
      const int kk = tid >> 3, mm = (tid & 7) * 8;
      const _Float16* src = Ah + (size_t)(kb + kk) * srcStride + (m0 + mm);
#pragma unroll
      for (int u = 0; u < 8; ++u) As[(mm + u) * ASTR + kk] = src[u];
      if (kb + BLK_K < K)
        __builtin_prefetch(src + (size_t)BLK_K * srcStride, 0, 1);
    }

    // ---- stage B tile (32 x 128) into LDS ----
    if constexpr (MODE == MODE_PV) {
      // B = P^T; P stored row-major [Ncols][K] with row stride srcStride
      const int j = tid >> 1, k0 = (tid & 1) * 16;
      const _Float16* src = Bh + (size_t)(n0 + j) * srcStride + kb + k0;
#pragma unroll
      for (int u = 0; u < 16; ++u) Bs[(k0 + u) * BSTR + j] = src[u];
      if (kb + BLK_K < K) __builtin_prefetch(src + BLK_K, 0, 1);
    } else {
      const int k = tid >> 3, j0 = (tid & 7) * 16;
      const _Float16* src = Bh + (size_t)(kb + k) * Ncols + n0 + j0;
      _Float16* dst = &Bs[k * BSTR + j0];
#pragma unroll
      for (int u = 0; u < 16; ++u) dst[u] = src[u];
      if (kb + BLK_K < K)
        __builtin_prefetch(src + (size_t)BLK_K * Ncols, 0, 1);
    }

    __syncthreads();

    // ---- fragments per ISA 16-bit layouts ----
    v16h afrag[2], bfrag[2];
#pragma unroll
    for (int ti = 0; ti < 2; ++ti) {
      const int row = wm * 32 + ti * 16 + l16;
      // A 16x32: lanes 0-15 K={0..7,16..23}, lanes 16-31 K={8..15,24..31}
      v8h alo = *(const v8h*)&As[row * ASTR + 8 * half];
      v8h ahi = *(const v8h*)&As[row * ASTR + 16 + 8 * half];
      afrag[ti] = __builtin_shufflevector(alo, ahi, 0, 1, 2, 3, 4, 5, 6, 7, 8,
                                          9, 10, 11, 12, 13, 14, 15);
    }
    const int kof = 8 * half;  // B 32x16: lane -> N col, K={0..7,16..23} / {8..15,24..31}
#pragma unroll
    for (int tj = 0; tj < 2; ++tj) {
      const int col = wn * 32 + tj * 16 + l16;
#pragma unroll
      for (int s = 0; s < 16; ++s)
        bfrag[tj][s] = Bs[(kof + (s & 7) + ((s >> 3) << 4)) * BSTR + col];
    }
#pragma unroll
    for (int ti = 0; ti < 2; ++ti)
#pragma unroll
      for (int tj = 0; tj < 2; ++tj)
        acc[ti][tj] = __builtin_amdgcn_wmma_f32_16x16x32_f16(
            false, afrag[ti], false, bfrag[tj], (short)0, acc[ti][tj], false,
            false);
  }

  // ---- epilogue: C/D layout -> lane N = lane%16, VGPR r -> M = r + 8*half ----
#pragma unroll
  for (int ti = 0; ti < 2; ++ti) {
#pragma unroll
    for (int tj = 0; tj < 2; ++tj) {
      const int gcol  = n0 + wn * 32 + tj * 16 + l16;
      const int growb = m0 + wm * 32 + ti * 16 + 8 * half;
#pragma unroll
      for (int r = 0; r < 8; ++r) {
        const int grow = growb + r;
        const float v = acc[ti][tj][r];
        const size_t o = (size_t)grow * Ncols + gcol;
        if constexpr (MODE == MODE_PROJ) {
          outh[o] = (_Float16)((v + bias[grow]) * scale);
        } else if constexpr (MODE == MODE_SCORES || MODE == MODE_PV) {
          outh[o] = (_Float16)v;
        } else {  // MODE_OUT: bias + residual, f32 store
          outf[o] = v + bias[grow] + resid[o];
        }
      }
    }
  }
}

// ---------------------------------------------------------------------------
// Host launcher. Workspace layout (halves):
//   hn  : B*C*HW       q/k/v : B*C*HW each
//   sb  : HW*HW  (per-batch scores, reused)     ao : C*HW (per-batch, reused)
// Total ~105 MB.
// ---------------------------------------------------------------------------
extern "C" void kernel_launch(void* const* d_in, const int* in_sizes, int n_in,
                              void* d_out, int out_size, void* d_ws,
                              size_t ws_size, hipStream_t stream)
{
  (void)in_sizes; (void)n_in; (void)out_size; (void)ws_size;
  const float* x     = (const float*)d_in[0];
  const float* gamma = (const float*)d_in[1];
  const float* beta  = (const float*)d_in[2];
  const float* wq = (const float*)d_in[3]; const float* bq = (const float*)d_in[4];
  const float* wk = (const float*)d_in[5]; const float* bk = (const float*)d_in[6];
  const float* wv = (const float*)d_in[7]; const float* bv = (const float*)d_in[8];
  const float* wo = (const float*)d_in[9]; const float* bo = (const float*)d_in[10];
  float* out = (float*)d_out;

  const size_t PLANE = (size_t)C_ * HW_;  // 2M elems
  _Float16* hn   = (_Float16*)d_ws;
  _Float16* qb   = hn + (size_t)B_ * PLANE;
  _Float16* kbuf = qb + (size_t)B_ * PLANE;
  _Float16* vb   = kbuf + (size_t)B_ * PLANE;
  _Float16* sb   = vb + (size_t)B_ * PLANE;    // HW_*HW_ halves
  _Float16* ao   = sb + (size_t)HW_ * HW_;     // PLANE halves

  groupnorm_kernel<<<B_ * G_, 1024, 0, stream>>>(x, gamma, beta, hn);

  const float scale = 0.044194173824159216f;  // 512^-0.5 folded into q
  dim3 blk(256);
  dim3 gProj(HW_ / BLK_N, C_ / BLK_M);    // 32 x 8
  dim3 gScore(HW_ / BLK_N, HW_ / BLK_M);  // 32 x 64

  for (int b = 0; b < B_; ++b) {
    const _Float16* hb = hn + (size_t)b * PLANE;
    wmma_gemm_kernel<MODE_PROJ><<<gProj, blk, 0, stream>>>(
        wq, nullptr, hb, bq, nullptr, qb + (size_t)b * PLANE, nullptr, scale,
        C_, HW_, C_, 0);
    wmma_gemm_kernel<MODE_PROJ><<<gProj, blk, 0, stream>>>(
        wk, nullptr, hb, bk, nullptr, kbuf + (size_t)b * PLANE, nullptr, 1.0f,
        C_, HW_, C_, 0);
    wmma_gemm_kernel<MODE_PROJ><<<gProj, blk, 0, stream>>>(
        wv, nullptr, hb, bv, nullptr, vb + (size_t)b * PLANE, nullptr, 1.0f,
        C_, HW_, C_, 0);
  }

  for (int b = 0; b < B_; ++b) {
    // scores = (q*scale)^T k  -> [4096 x 4096] f16 (per-batch, L2 resident)
    wmma_gemm_kernel<MODE_SCORES><<<gScore, blk, 0, stream>>>(
        nullptr, qb + (size_t)b * PLANE, kbuf + (size_t)b * PLANE, nullptr,
        nullptr, sb, nullptr, 1.0f, HW_, HW_, C_, HW_);
    softmax_kernel<<<HW_, 256, 0, stream>>>(sb);
    // ao = V * P^T  -> [512 x 4096] f16
    wmma_gemm_kernel<MODE_PV><<<gProj, blk, 0, stream>>>(
        nullptr, vb + (size_t)b * PLANE, sb, nullptr, nullptr, ao, nullptr,
        1.0f, C_, HW_, HW_, HW_);
    // out = Wo * ao + bo + x (residual), f32
    wmma_gemm_kernel<MODE_OUT><<<gProj, blk, 0, stream>>>(
        wo, nullptr, ao, bo, x + (size_t)b * PLANE, nullptr,
        out + (size_t)b * PLANE, 1.0f, C_, HW_, C_, 0);
  }
}
